// corrs2_893353197606
// MI455X (gfx1250) — matrix operand order repaired and checked by reading, not compiled, not run on hardware
//
#include <hip/hip_runtime.h>
#include <hip/hip_bf16.h>

// ---------------------------------------------------------------------------
// Problem constants (from reference)
// ---------------------------------------------------------------------------
#define BB   4
#define CC   128
#define HH   512
#define WW0  512
#define hh   128          // H/4
#define ww   128          // W/4
#define KK   4
#define GG   4
#define CG   32           // CC / GG
#define HP   136          // hh + 2K
#define WP   136
#define HP2  138          // HP + 2 (zero ring for conv pad=1)
#define WP2  138
#define NOFF 81
#define EPSF 1e-8f

typedef __attribute__((ext_vector_type(16))) __bf16 v16bf;
typedef __attribute__((ext_vector_type(8)))  float  v8f;

// align-corners scale 511/127
#define SCALEF (511.0f / 127.0f)

// Channel permutation inside each 32-channel chunk: swap bits 3<->4.
// Storing yp with this order makes each lane's 16-element WMMA A-fragment
// one contiguous 32B region:
//   hi=0 lanes read pos [0..15]  = channels {0..7, 16..23}
//   hi=1 lanes read pos [16..31] = channels {8..15, 24..31}
// which is exactly the ISA 16-bit A-matrix per-lane K layout.
__device__ __forceinline__ int swap34(int c) {
    return (c & 7) | ((c & 8) << 1) | ((c & 16) >> 1);
}

__device__ __forceinline__ int reflect_idx(int i) {
    // jnp.pad 'reflect' (no edge repeat), size hh, pad KK (KK < hh)
    if (i < 0)   i = -i;
    if (i >= hh) i = 2 * (hh - 1) - i;
    return i;
}

__device__ __forceinline__ float bilerp_src(const float* src, int b, int c,
                                            int sy, int sx) {
    // sample resized-(128) pixel (sy,sx) from original 512x512 NCHW plane
    float fy = sy * SCALEF;
    float fx = sx * SCALEF;
    int y0 = (int)floorf(fy);
    int x0 = (int)floorf(fx);
    int y1 = min(y0 + 1, HH - 1);
    int x1 = min(x0 + 1, WW0 - 1);
    float wy = fy - (float)y0;
    float wx = fx - (float)x0;
    const float* p = src + ((size_t)(b * CC + c) * HH) * WW0;
    float a00 = p[(size_t)y0 * WW0 + x0];
    float a01 = p[(size_t)y0 * WW0 + x1];
    float a10 = p[(size_t)y1 * WW0 + x0];
    float a11 = p[(size_t)y1 * WW0 + x1];
    float top = a00 * (1.0f - wx) + a01 * wx;
    float bot = a10 * (1.0f - wx) + a11 * wx;
    return top * (1.0f - wy) + bot * wy;
}

// ---------------------------------------------------------------------------
// K1: downsample x -> xs, NHWC f32  [B, hh, ww, CC] (natural channel order)
// ---------------------------------------------------------------------------
__global__ void k_resize_x(const float* __restrict__ x, float* __restrict__ xs) {
    int idx = blockIdx.x * blockDim.x + threadIdx.x;
    if (idx >= BB * CC * hh * ww) return;
    int px = idx % ww;
    int py = (idx / ww) % hh;
    int c  = (idx / (ww * hh)) % CC;
    int b  = idx / (ww * hh * CC);
    float v = bilerp_src(x, b, c, py, px);
    xs[(((size_t)(b * hh + py) * ww) + px) * CC + c] = v;
}

// ---------------------------------------------------------------------------
// K2: downsample y + reflect-pad(4) + zero ring(1) -> yp, NHWC bf16
//     [B, HP2, WP2, CC]; channels stored in swap34-permuted order so that
//     conv A-fragments are contiguous.  Permutation is intra-64B, so the
//     coalesced write pattern is unchanged.
// ---------------------------------------------------------------------------
__global__ void k_resize_pad_y(const float* __restrict__ y, __bf16* __restrict__ yp) {
    int idx = blockIdx.x * blockDim.x + threadIdx.x;
    if (idx >= BB * HP2 * WP2 * CC) return;
    int c  = idx % CC;
    int xx = (idx / CC) % WP2;
    int yy = (idx / (CC * WP2)) % HP2;
    int b  = idx / (CC * WP2 * HP2);
    float v = 0.0f;
    if (yy >= 1 && yy <= HP && xx >= 1 && xx <= WP) {
        int sy = reflect_idx((yy - 1) - KK);
        int sx = reflect_idx((xx - 1) - KK);
        v = bilerp_src(y, b, c, sy, sx);
    }
    int cpos = (c & 96) | swap34(c & 31);
    yp[(((size_t)(b * HP2 + yy) * WP2) + xx) * CC + cpos] = (__bf16)v;
}

// ---------------------------------------------------------------------------
// K3: pre-swizzle W (OIHW f32) into WMMA B-fragments, bf16.
//     B-matrix (32x16 bf16) per-lane layout (mirror of ISA A layout):
//       lane l<16  : N = l,      elements e: K = {0..7, 16..23}[e]
//       lane l>=16 : N = l - 16, elements e: K = {8..15, 24..31}[e]
//     Fragment buffer: [tap(9)][kchunk(4)][ntile(8)][lane(32)][elem(16)]
// ---------------------------------------------------------------------------
__global__ void k_prep_wfrag(const float* __restrict__ W, __bf16* __restrict__ wf) {
    int idx = blockIdx.x * blockDim.x + threadIdx.x;
    if (idx >= 9 * 4 * 8 * 32 * 16) return;
    int e  = idx % 16;
    int l  = (idx / 16) % 32;
    int nt = (idx / (16 * 32)) % 8;
    int kc = (idx / (16 * 32 * 8)) % 4;
    int t  = idx / (16 * 32 * 8 * 4);
    int hi = l >> 4;
    int n  = nt * 16 + (l & 15);
    int kl = (e < 8) ? (hi * 8 + e) : (16 + hi * 8 + (e - 8));
    int cin = kc * 32 + kl;
    int di = t / 3, dj = t % 3;
    float v = W[(((size_t)n * CC + cin) * 3 + di) * 3 + dj];
    wf[idx] = (__bf16)v;
}

// ---------------------------------------------------------------------------
// K4: implicit-GEMM 3x3 conv with v_wmma_f32_16x16x32_bf16.
//     One block (8 waves) = one (b, row, 16-pixel tile); the block stages the
//     3x18x128ch bf16 activation patch (13.8 KB) into LDS once, then each
//     wave computes one 16(pixel) x 16(out-channel) tile:
//       9 taps x 4 chunks = 36 WMMAs, A from LDS (contiguous 32B/lane),
//       B from the L2-hot pre-swizzled weight fragments.
//     Output: yc NHWC f32 [B, HP, WP, CC] (+bias), natural channel order.
// ---------------------------------------------------------------------------
#define PXT 9                 // ceil(136/16)
#define SM_W 18               // staged columns: 16 pixels + 2 halo
#define SM_ELEMS (3 * SM_W * CC)

__global__ void __launch_bounds__(256)
k_conv_wmma(const __bf16* __restrict__ yp,
            const __bf16* __restrict__ wf,
            const float*  __restrict__ bconv,
            float* __restrict__ yc) {
    __shared__ alignas(16) __bf16 smem[SM_ELEMS];   // 13824 B

    int grp = blockIdx.x;                 // 0 .. BB*HP*PXT-1
    int pxt = grp % PXT;
    int py  = (grp / PXT) % HP;
    int b   = grp / (PXT * HP);
    int px_base = pxt * 16;

    int nt   = threadIdx.x >> 5;          // wave id = output-channel tile
    int lane = threadIdx.x & 31;
    int m    = lane & 15;
    int hi   = lane >> 4;

    // ---- cooperative stage: rows py..py+2, cols px_base..px_base+17 ----
    for (int ch = threadIdx.x; ch < SM_ELEMS / 8; ch += 256) {
        int base = ch * 8;
        int r    = base / (SM_W * CC);
        int rem  = base % (SM_W * CC);
        int col  = rem / CC;
        int c0   = rem % CC;
        int yy = py + r;                             // 0..137
        int xx = px_base + col;
        if (xx > WP2 - 1) xx = WP2 - 1;              // clamp partial tile
        const __bf16* src = yp + (((size_t)(b * HP2 + yy) * WP2) + xx) * CC + c0;
        *(uint4*)(&smem[base]) = *(const uint4*)src; // 16B chunks
    }
    __syncthreads();

    // ---- 36 WMMAs ----
    v8f acc = {};
    const __bf16* wfl = wf + ((size_t)nt * 32 + lane) * 16;
    #pragma unroll
    for (int t = 0; t < 9; ++t) {
        int di = t / 3, dj = t % 3;
        const __bf16* arow = smem + (di * SM_W + (m + dj)) * CC;
        #pragma unroll
        for (int kc = 0; kc < 4; ++kc) {
            v16bf a   = *(const v16bf*)(arow + kc * 32 + hi * 16); // ds_load x2
            v16bf bfr = *(const v16bf*)(wfl + (size_t)(t * 4 + kc) * (8 * 32 * 16));
            acc = __builtin_amdgcn_wmma_f32_16x16x32_bf16(
                      false, a, false, bfr, (short)0, acc, false, false);
        }
    }

    // D layout: VGPR r, lane<16 -> (M=r, N=lane); lane>=16 -> (M=r+8, N=lane-16)
    int nglob = nt * 16 + (lane & 15);
    float bias = bconv[nglob];
    #pragma unroll
    for (int r = 0; r < 8; ++r) {
        int pxx = px_base + r + hi * 8;
        if (pxx < WP) {
            yc[(((size_t)(b * HP + py) * WP) + pxx) * CC + nglob] = acc[r] + bias;
        }
    }
}

// ---------------------------------------------------------------------------
// K5a/K5b: per-group L2 norms
// ---------------------------------------------------------------------------
__global__ void k_norm_x(const float* __restrict__ xs, float* __restrict__ nx) {
    int idx = blockIdx.x * blockDim.x + threadIdx.x;
    if (idx >= BB * GG * hh * ww) return;
    int px = idx % ww;
    int py = (idx / ww) % hh;
    int g  = (idx / (ww * hh)) % GG;
    int b  = idx / (ww * hh * GG);
    const float* p = xs + (((size_t)(b * hh + py) * ww) + px) * CC + g * CG;
    float s = 0.0f;
    #pragma unroll
    for (int i = 0; i < CG / 4; ++i) {
        float4 v = ((const float4*)p)[i];
        s += v.x * v.x + v.y * v.y + v.z * v.z + v.w * v.w;
    }
    nx[idx] = fmaxf(sqrtf(s), EPSF);
}

__global__ void k_norm_y(const float* __restrict__ yc, float* __restrict__ ny) {
    int idx = blockIdx.x * blockDim.x + threadIdx.x;
    if (idx >= BB * GG * HP * WP) return;
    int px = idx % WP;
    int py = (idx / WP) % HP;
    int g  = (idx / (WP * HP)) % GG;
    int b  = idx / (WP * HP * GG);
    const float* p = yc + (((size_t)(b * HP + py) * WP) + px) * CC + g * CG;
    float s = 0.0f;
    #pragma unroll
    for (int i = 0; i < CG / 4; ++i) {
        float4 v = ((const float4*)p)[i];
        s += v.x * v.x + v.y * v.y + v.z * v.z + v.w * v.w;
    }
    ny[idx] = sqrtf(s);
}

// ---------------------------------------------------------------------------
// K6: 81-offset normalized correlation. Output (B, G, 81, hh, ww) f32.
//     One thread per (b,g,py,px); x vector kept in registers (8 x float4),
//     81 y-window dots from L2-resident yc. Output writes coalesced over px.
// ---------------------------------------------------------------------------
__global__ void k_corr(const float* __restrict__ xs,
                       const float* __restrict__ yc,
                       const float* __restrict__ nx,
                       const float* __restrict__ ny,
                       float* __restrict__ out) {
    int idx = blockIdx.x * blockDim.x + threadIdx.x;
    if (idx >= BB * GG * hh * ww) return;
    int px = idx & (ww - 1);
    int py = (idx >> 7) & (hh - 1);
    int g  = (idx >> 14) & (GG - 1);
    int b  = idx >> 16;

    const float* xp = xs + (((size_t)(b * hh + py) * ww) + px) * CC + g * CG;
    float4 xv[8];
    #pragma unroll
    for (int i = 0; i < 8; ++i) xv[i] = ((const float4*)xp)[i];
    float nxv = nx[idx];

    size_t obase = (((size_t)(b * GG + g) * NOFF) * hh) * ww
                 + (size_t)py * ww + px;

    for (int o = 0; o < NOFF; ++o) {
        int di = o / 9, dj = o % 9;
        const float* yptr =
            yc + (((size_t)(b * HP + py + di) * WP) + (px + dj)) * CC + g * CG;
        __builtin_prefetch(yptr, 0, 0);
        float dot = 0.0f;
        #pragma unroll
        for (int i = 0; i < 8; ++i) {
            float4 v = ((const float4*)yptr)[i];
            dot += xv[i].x * v.x + xv[i].y * v.y + xv[i].z * v.z + xv[i].w * v.w;
        }
        float nyv = ny[((size_t)(b * GG + g) * HP + (py + di)) * WP + (px + dj)];
        out[obase + (size_t)o * hh * ww] = dot / (nxv * fmaxf(nyv, EPSF));
    }
}

// ---------------------------------------------------------------------------
// Launcher
// ---------------------------------------------------------------------------
extern "C" void kernel_launch(void* const* d_in, const int* in_sizes, int n_in,
                              void* d_out, int out_size, void* d_ws, size_t ws_size,
                              hipStream_t stream) {
    const float* x     = (const float*)d_in[0];
    const float* y     = (const float*)d_in[1];
    const float* W     = (const float*)d_in[2];
    const float* bconv = (const float*)d_in[3];
    float* out = (float*)d_out;

    char* ws = (char*)d_ws;
    size_t off = 0;
    float*  xs = (float*)(ws + off);  off += (size_t)BB * hh * ww * CC * 4;     // 32 MB
    __bf16* yp = (__bf16*)(ws + off); off += (size_t)BB * HP2 * WP2 * CC * 2;   // 19.5 MB
    __bf16* wf = (__bf16*)(ws + off); off += (size_t)9 * 4 * 8 * 32 * 16 * 2;   // 288 KB
    off = (off + 255) & ~(size_t)255;
    float*  yc = (float*)(ws + off);  off += (size_t)BB * HP * WP * CC * 4;     // 37.9 MB
    float*  nx = (float*)(ws + off);  off += (size_t)BB * GG * hh * ww * 4;     // 1 MB
    float*  ny = (float*)(ws + off);  off += (size_t)BB * GG * HP * WP * 4;     // 1.2 MB

    const int T = 256;

    int n1 = BB * CC * hh * ww;
    k_resize_x<<<(n1 + T - 1) / T, T, 0, stream>>>(x, xs);

    int n2 = BB * HP2 * WP2 * CC;
    k_resize_pad_y<<<(n2 + T - 1) / T, T, 0, stream>>>(y, yp);

    int n3 = 9 * 4 * 8 * 32 * 16;
    k_prep_wfrag<<<(n3 + T - 1) / T, T, 0, stream>>>(W, wf);

    int nblk = BB * HP * PXT;             // one block per pixel-tile group
    k_conv_wmma<<<nblk, T, 0, stream>>>(yp, wf, bconv, yc);

    int n5 = BB * GG * hh * ww;
    k_norm_x<<<(n5 + T - 1) / T, T, 0, stream>>>(xs, nx);
    int n6 = BB * GG * HP * WP;
    k_norm_y<<<(n6 + T - 1) / T, T, 0, stream>>>(yc, ny);

    k_corr<<<(n5 + T - 1) / T, T, 0, stream>>>(xs, yc, nx, ny, out);
}
